// IHGNN_29240137351497
// MI455X (gfx1250) — matrix operand order repaired and checked by reading, not compiled, not run on hardware
//
#include <hip/hip_runtime.h>
#include <hip/hip_bf16.h>

// -------------------- types --------------------
typedef __attribute__((ext_vector_type(16))) __bf16 v16bf;
typedef __attribute__((ext_vector_type(8)))  float  v8f;

#define WMMA_BF16(A, B, C) \
  __builtin_amdgcn_wmma_f32_16x16x32_bf16(false, (A), false, (B), (short)0, (C), false, false)

// -------------------- WMMA fragment helpers --------------------
// A-fragment (16x32, bf16) from f32 row-major [*, ld], rows row0..row0+15, K offset k0.
// Lane L<16 : row M=L,    K = k0+0..15
// Lane L>=16: row M=L-16, K = k0+16..31
static __device__ __forceinline__ v16bf load_a_f32(const float* __restrict__ src,
                                                   int row0, int ld, int k0) {
  int lane = threadIdx.x & 31;
  int m    = lane & 15;
  int h8   = ((lane >> 4) & 1) * 16;
  const float* p = src + (size_t)(row0 + m) * ld + k0;
  v16bf a;
#pragma unroll
  for (int i = 0; i < 16; ++i) a[i]     = (__bf16)p[h8 + i];
  return a;
}

// A-fragment from bf16 LDS tile [16][32]
static __device__ __forceinline__ v16bf load_a_lds(const __bf16* __restrict__ h) {
  int lane = threadIdx.x & 31;
  int m    = lane & 15;
  int h8   = ((lane >> 4) & 1) * 16;
  const __bf16* p = h + m * 32;
  v16bf a;
#pragma unroll
  for (int i = 0; i < 16; ++i) a[i]     = p[h8 + i];
  return a;
}

// B-fragment (32x16, bf16) from f32 row-major weight [K][32], K rows k0..k0+31, cols n0..n0+15.
// Lane L<16 : col N=L,    K=k0+0..15 ; Lane L>=16: col N=L-16, K=k0+16..31
static __device__ __forceinline__ v16bf load_b_f32(const float* __restrict__ w,
                                                   int k0, int n0) {
  int lane = threadIdx.x & 31;
  int col  = n0 + (lane & 15);
  int kh   = ((lane >> 4) & 1) * 16;
  v16bf b;
#pragma unroll
  for (int i = 0; i < 16; ++i) b[i] = (__bf16)w[(size_t)(k0 + kh + i) * 32 + col];
  return b;
}

// Combined B-fragment: rows (k0a + k) + (k0b + k) of a [96][32] weight (concat elimination).
static __device__ __forceinline__ v16bf load_b_comb(const float* __restrict__ w,
                                                    int k0a, int k0b, int n0) {
  int lane = threadIdx.x & 31;
  int col  = n0 + (lane & 15);
  int kh   = ((lane >> 4) & 1) * 16;
  v16bf b;
#pragma unroll
  for (int i = 0; i < 16; ++i) {
    int k = kh + i;
    b[i] = (__bf16)(w[(size_t)(k0a + k) * 32 + col] + w[(size_t)(k0b + k) * 32 + col]);
  }
  return b;
}

// Load a pre-packed B-fragment: frags[fragidx][lane][16 bf16] -> one 32B/lane vector load.
static __device__ __forceinline__ v16bf load_frag(const __bf16* __restrict__ frags, int fragidx) {
  int lane = threadIdx.x & 31;
  return *(const v16bf*)(frags + (size_t)fragidx * 512 + (size_t)lane * 16);
}

// -------------------- kernel 0: pre-pack all weight B-fragments (bf16, lane layout) ----------
// frag ids: 0..7  = w0a (kc = id>>1, n0 = (id&1)*16)
//           8..9  = w0b (n0 = (id-8)*16)
//           10+   = per layer l, 6 frags: [Wea n0=0/16, Wna n0=0/16, W1b n0=0/16]
__global__ void __launch_bounds__(32)
ihgnn_prep(const float* __restrict__ w0a, const float* __restrict__ w0b,
           const float* __restrict__ W1a, const float* __restrict__ W1b,
           __bf16* __restrict__ frags) {
  const int id   = blockIdx.x;
  const int lane = threadIdx.x & 31;
  v16bf b;
  if (id < 8) {
    b = load_b_f32(w0a, (id >> 1) * 32, (id & 1) * 16);
  } else if (id < 10) {
    b = load_b_f32(w0b, 0, (id - 8) * 16);
  } else {
    int l = (id - 10) / 6, j = (id - 10) % 6;
    const float* Wa = W1a + (size_t)l * 96 * 32;
    const float* Wb = W1b + (size_t)l * 32 * 32;
    if (j < 2)      b = load_b_comb(Wa, 0, 64, (j & 1) * 16);   // Wea = Wa0 + Wa2
    else if (j < 4) b = load_b_comb(Wa, 32, 64, (j & 1) * 16);  // Wna = Wa1 + Wa2
    else            b = load_b_f32(Wb, 0, (j & 1) * 16);
  }
  *(v16bf*)(frags + (size_t)id * 512 + (size_t)lane * 16) = b;
}

// -------------------- kernel 1: ego0 = MLP(node_feat); out = alpha0*ego0 --------------------
__global__ void __launch_bounds__(128)
ihgnn_mlp0(const float* __restrict__ X, const __bf16* __restrict__ frags,
           const float* __restrict__ b0a, const float* __restrict__ b0b,
           const float* __restrict__ alpha,
           float* __restrict__ ego, float* __restrict__ outacc) {
  __shared__ __bf16 lds_h[4][16 * 32];
  const int wave = threadIdx.x >> 5;
  const int lane = threadIdx.x & 31;
  const int row0 = (blockIdx.x * 4 + wave) * 16;

  v8f z = {};
  v8f h0 = z, h1 = z;
#pragma unroll
  for (int kc = 0; kc < 4; ++kc) {
    v16bf ax = load_a_f32(X, row0, 128, kc * 32);
    v16bf b0 = load_frag(frags, kc * 2 + 0);
    v16bf b1 = load_frag(frags, kc * 2 + 1);
    h0 = WMMA_BF16(ax, b0, h0);
    h1 = WMMA_BF16(ax, b1, h1);
  }

  // bias + ReLU, re-layout via LDS (C-frag -> A-frag)
  {
    int n  = lane & 15;
    int mh = ((lane >> 4) & 1) * 8;
    float ba0 = b0a[n], ba1 = b0a[16 + n];
    __bf16* h = lds_h[wave];
#pragma unroll
    for (int r = 0; r < 8; ++r) {
      float x0 = h0[r] + ba0; x0 = fmaxf(x0, 0.f);
      float x1 = h1[r] + ba1; x1 = fmaxf(x1, 0.f);
      h[(mh + r) * 32 + n]      = (__bf16)x0;
      h[(mh + r) * 32 + 16 + n] = (__bf16)x1;
    }
  }
  __syncthreads();

  v16bf ah  = load_a_lds(lds_h[wave]);
  v16bf wb0 = load_frag(frags, 8);
  v16bf wb1 = load_frag(frags, 9);
  v8f e0 = WMMA_BF16(ah, wb0, z);
  v8f e1 = WMMA_BF16(ah, wb1, z);

  const float a0 = alpha[0];
  int n  = lane & 15;
  int mh = ((lane >> 4) & 1) * 8;
  float bb0 = b0b[n], bb1 = b0b[16 + n];
#pragma unroll
  for (int r = 0; r < 8; ++r) {
    size_t row = (size_t)(row0 + mh + r);
    float x0 = e0[r] + bb0;
    float x1 = e1[r] + bb1;
    ego[row * 32 + n]         = x0;
    ego[row * 32 + 16 + n]    = x1;
    outacc[row * 32 + n]      = a0 * x0;
    outacc[row * 32 + 16 + n] = a0 * x1;
  }
}

// -------------------- zero + edge scatter (SpMM) --------------------
__global__ void ihgnn_zero(float* __restrict__ p, int n) {
  int i = blockIdx.x * blockDim.x + threadIdx.x;
  if (i < n) p[i] = 0.f;
}

__global__ void ihgnn_scatter(const float* __restrict__ ego, const int* __restrict__ src,
                              const int* __restrict__ dst, float* __restrict__ neig, int E) {
  int gid = blockIdx.x * blockDim.x + threadIdx.x;  // 32 lanes per edge
  int e = gid >> 5, f = gid & 31;
  if (e < E) {
    float v = ego[(size_t)src[e] * 32 + f];
    atomicAdd(&neig[(size_t)dst[e] * 32 + f], v);
  }
}

// -------------------- kernel 3: fused layer (concat-free GEMM + accumulate) --------------------
// hidden = relu(ego@Wea + neig@Wna + B1a) ; ego' = hidden@W1b + B1b ; out += alpha*ego'
__global__ void __launch_bounds__(128)
ihgnn_layer(const float* __restrict__ ego_in, const float* __restrict__ neig,
            const __bf16* __restrict__ frags,  // 6 frags for this layer
            const float* __restrict__ B1a, const float* __restrict__ B1b,
            const float* __restrict__ alpha, int al_idx,
            float* __restrict__ ego_out, float* __restrict__ outacc) {
  __shared__ __bf16 lds_h[4][16 * 32];
  const int wave = threadIdx.x >> 5;
  const int lane = threadIdx.x & 31;
  const int row0 = (blockIdx.x * 4 + wave) * 16;

  v16bf wea0 = load_frag(frags, 0);
  v16bf wea1 = load_frag(frags, 1);
  v16bf wna0 = load_frag(frags, 2);
  v16bf wna1 = load_frag(frags, 3);
  v16bf wb0  = load_frag(frags, 4);
  v16bf wb1  = load_frag(frags, 5);

  v16bf ae = load_a_f32(ego_in, row0, 32, 0);
  v16bf an = load_a_f32(neig,   row0, 32, 0);

  v8f z = {};
  v8f h0 = WMMA_BF16(ae, wea0, z);
  h0     = WMMA_BF16(an, wna0, h0);
  v8f h1 = WMMA_BF16(ae, wea1, z);
  h1     = WMMA_BF16(an, wna1, h1);

  {
    int n  = lane & 15;
    int mh = ((lane >> 4) & 1) * 8;
    float ba0 = B1a[n], ba1 = B1a[16 + n];
    __bf16* h = lds_h[wave];
#pragma unroll
    for (int r = 0; r < 8; ++r) {
      float x0 = h0[r] + ba0; x0 = fmaxf(x0, 0.f);
      float x1 = h1[r] + ba1; x1 = fmaxf(x1, 0.f);
      h[(mh + r) * 32 + n]      = (__bf16)x0;
      h[(mh + r) * 32 + 16 + n] = (__bf16)x1;
    }
  }
  __syncthreads();

  v16bf ah = load_a_lds(lds_h[wave]);
  v8f e0 = WMMA_BF16(ah, wb0, z);
  v8f e1 = WMMA_BF16(ah, wb1, z);

  const float a = alpha[al_idx];
  int n  = lane & 15;
  int mh = ((lane >> 4) & 1) * 8;
  float bb0 = B1b[n], bb1 = B1b[16 + n];
#pragma unroll
  for (int r = 0; r < 8; ++r) {
    size_t row = (size_t)(row0 + mh + r);
    float x0 = e0[r] + bb0;
    float x1 = e1[r] + bb1;
    ego_out[row * 32 + n]      = x0;
    ego_out[row * 32 + 16 + n] = x1;
    outacc[row * 32 + n]      += a * x0;
    outacc[row * 32 + 16 + n] += a * x1;
  }
}

// -------------------- kernel 4: per-graph top-k (k=30) on wl_color + gather + ReLU -----------
__global__ void __launch_bounds__(512)
ihgnn_topk(const float* __restrict__ ego, const float* __restrict__ outacc,
           float* __restrict__ out, int NPG, int K) {
  __shared__ float v[512];
  __shared__ float rv[512];
  __shared__ int   ri[512];
  __shared__ int   sel[32];
  const int g = blockIdx.x;
  const int t = threadIdx.x;

  float val = -__builtin_huge_valf();
  if (t < NPG) val = ego[((size_t)g * NPG + t) * 32 + 31];  // wl_color = last channel
  v[t] = val;
  __syncthreads();

  for (int k = 0; k < K; ++k) {
    rv[t] = v[t];
    ri[t] = t;
    __syncthreads();
#pragma unroll
    for (int s = 256; s > 0; s >>= 1) {
      if (t < s) {
        float a = rv[t], b = rv[t + s];
        int ia = ri[t], ib = ri[t + s];
        if (b > a || (b == a && ib < ia)) { rv[t] = b; ri[t] = ib; }
      }
      __syncthreads();
    }
    if (t == 0) { sel[k] = ri[0]; v[ri[0]] = -__builtin_huge_valf(); }
    __syncthreads();
  }

  const int KD = K * 32;
  for (int j = t; j < KD; j += 512) {
    int ki = j >> 5, f = j & 31;
    float x = outacc[((size_t)g * NPG + sel[ki]) * 32 + f];
    out[(size_t)g * KD + j] = fmaxf(x, 0.f);
  }
}

// -------------------- host side --------------------
extern "C" void kernel_launch(void* const* d_in, const int* in_sizes, int n_in,
                              void* d_out, int out_size, void* d_ws, size_t ws_size,
                              hipStream_t stream) {
  const float* node_feat = (const float*)d_in[0];
  const float* alpha     = (const float*)d_in[1];
  const float* w0a       = (const float*)d_in[2];
  const float* b0a       = (const float*)d_in[3];
  const float* w0b       = (const float*)d_in[4];
  const float* b0b       = (const float*)d_in[5];
  const float* W1a       = (const float*)d_in[6];
  const float* B1a       = (const float*)d_in[7];
  const float* W1b       = (const float*)d_in[8];
  const float* B1b       = (const float*)d_in[9];
  const int*   esrc      = (const int*)d_in[10];
  const int*   edst      = (const int*)d_in[11];

  const int F = 128, D = 32, K = 30, L = 3;
  const int N = in_sizes[0] / F;              // 128000
  const int E = in_sizes[10];                 // 2048000
  const int B = out_size / (K * D);           // 256
  const int NPG = N / B;                      // 500

  float*  ego    = (float*)d_ws;              // [N, D]
  float*  neig   = ego  + (size_t)N * D;      // [N, D]
  float*  outacc = neig + (size_t)N * D;      // [N, D]
  __bf16* frags  = (__bf16*)(outacc + (size_t)N * D);  // 28 frags * 512 bf16

  const int nfrags = 10 + 6 * L;              // 28
  ihgnn_prep<<<nfrags, 32, 0, stream>>>(w0a, w0b, W1a, W1b, frags);

  // ego0 = MLP(node_feat); out = alpha0*ego0
  ihgnn_mlp0<<<N / 64, 128, 0, stream>>>(node_feat, frags, b0a, b0b, alpha, ego, outacc);

  for (int l = 0; l < L; ++l) {
    const int nz = N * D;
    ihgnn_zero<<<(nz + 255) / 256, 256, 0, stream>>>(neig, nz);
    const long long tot = (long long)E * D;
    ihgnn_scatter<<<(int)((tot + 255) / 256), 256, 0, stream>>>(ego, esrc, edst, neig, E);
    ihgnn_layer<<<N / 64, 128, 0, stream>>>(ego, neig, frags + (size_t)(10 + 6 * l) * 512,
                                            B1a + l * D, B1b + l * D,
                                            alpha, l + 1, ego, outacc);
  }

  ihgnn_topk<<<B, 512, 0, stream>>>(ego, outacc, (float*)d_out, NPG, K);
}
